// CausalSelfAttention_32547262169356
// MI455X (gfx1250) — matrix operand order repaired and checked by reading, not compiled, not run on hardware
//
#include <hip/hip_runtime.h>
#include <math.h>

// ---------------------------------------------------------------------------
// CDNA5 (gfx1250) causal self-attention: bf16 WMMA + async LDS staging.
// ---------------------------------------------------------------------------

typedef __attribute__((ext_vector_type(16))) __bf16 v16bf;
typedef __attribute__((ext_vector_type(8)))  __bf16 v8bf;
typedef __attribute__((ext_vector_type(8)))  float  v8f;

#define AS3 __attribute__((address_space(3)))

// Async global->LDS staging via inline asm (ISA 15.18.3, GV mode):
//   global_load_async_to_lds_b128 vdst(LDS byte addr), vaddr(64b global VA), off
#define USE_ASYNC_LDS_ASM 1

__device__ __forceinline__ v8f wmma_bf16(v16bf a, v16bf b, v8f c) {
  return __builtin_amdgcn_wmma_f32_16x16x32_bf16(
      false, a, false, b, (short)0, c, false, false);
}

// 16-byte global -> LDS copy (async path overlaps with WMMA compute)
__device__ __forceinline__ void copy16B_g2l(const __bf16* g, __bf16* l) {
#if USE_ASYNC_LDS_ASM
  const unsigned laddr = (unsigned)(size_t)(AS3 __bf16*)l;   // LDS byte address
  const unsigned long long gaddr = (unsigned long long)(size_t)g;
  asm volatile("global_load_async_to_lds_b128 %0, %1, off"
               :
               : "v"(laddr), "v"(gaddr)
               : "memory");
#else
  *(v8bf*)l = *(const v8bf*)g;
#endif
}

__device__ __forceinline__ void wait_stage_and_sync() {
#if USE_ASYNC_LDS_ASM
  asm volatile("s_wait_asynccnt 0" ::: "memory");
#endif
  __syncthreads();
}

// ---------------------------------------------------------------------------
// fp32 -> bf16 elementwise convert
// ---------------------------------------------------------------------------
__global__ void cvt_f32_to_bf16(const float* __restrict__ in,
                                __bf16* __restrict__ out, int n) {
  int i = blockIdx.x * blockDim.x + threadIdx.x;
  if (i < n) out[i] = (__bf16)in[i];
}

// ---------------------------------------------------------------------------
// GEMM: Y = A[M,K] * W[N,K]^T, bf16 in, f32 accumulate.
// One wave -> 64x64 output tile: 16 accumulators, 16 WMMA per 32-wide k-step
// vs 16 b128 loads -> VMEM and WMMA pipes roughly balanced.
// mode 0: bf16 head-major [B,H,S,Dk]   (Q, K)
// mode 1: f32 row-major [M,N]          (final projection)
// mode 2: bf16 transposed head-major [B,H,Dk,S]  (V^T for PV fragments)
// ---------------------------------------------------------------------------
__global__ __launch_bounds__(256)
void gemm_bf16_wmma(const __bf16* __restrict__ A, const __bf16* __restrict__ W,
                    void* __restrict__ out, int M, int N, int K, int mode) {
  const int lane = threadIdx.x & 31;
  const int l16  = lane & 15;
  const int half = lane >> 4;
  const int wave = blockIdx.x * 8 + (threadIdx.x >> 5);
  const int ngroups = N >> 6;
  const int mt = wave / ngroups;        // 64-row tile index
  const int ng = wave % ngroups;        // 64-col group
  if (mt * 64 >= M) return;             // wave-uniform

  const __bf16* Abase = A + (size_t)(mt * 64 + l16) * K;   // + sub*16*K
  const __bf16* Wbase = W + (size_t)(ng * 64 + l16) * K;   // + t*16*K

  v8f c[4][4];
#pragma unroll
  for (int s = 0; s < 4; ++s)
#pragma unroll
    for (int t = 0; t < 4; ++t) c[s][t] = v8f{};

  for (int k0 = 0; k0 < K; k0 += 32) {
    v16bf b[4];
#pragma unroll
    for (int t = 0; t < 4; ++t)
      b[t] = *(const v16bf*)(Wbase + (size_t)t * 16 * K + k0 + half * 16);
    v16bf a[4];
#pragma unroll
    for (int s = 0; s < 4; ++s) {
      v8bf lo = *(const v8bf*)(Abase + (size_t)s * 16 * K + k0 + half * 16);
      v8bf hi = *(const v8bf*)(Abase + (size_t)s * 16 * K + k0 + half * 16 + 8);
#pragma unroll
      for (int i = 0; i < 8; ++i) { a[s][i] = lo[i]; a[s][8 + i] = hi[i]; }
    }
#pragma unroll
    for (int s = 0; s < 4; ++s)
#pragma unroll
      for (int t = 0; t < 4; ++t)
        c[s][t] = wmma_bf16(a[s], b[t], c[s][t]);
  }

#pragma unroll
  for (int s = 0; s < 4; ++s) {
#pragma unroll
    for (int t = 0; t < 4; ++t) {
#pragma unroll
      for (int r = 0; r < 8; ++r) {
        const int row = mt * 64 + s * 16 + half * 8 + r;
        const int col = ng * 64 + t * 16 + l16;
        const float v = c[s][t][r];
        const int bi = row >> 11, ss = row & 2047;
        const int h  = col >> 6,  d  = col & 63;
        if (mode == 0) {
          ((__bf16*)out)[(((size_t)bi * 16 + h) * 2048 + ss) * 64 + d] = (__bf16)v;
        } else if (mode == 2) {
          ((__bf16*)out)[(((size_t)bi * 16 + h) * 64 + d) * 2048 + ss] = (__bf16)v;
        } else {
          ((float*)out)[(size_t)row * N + col] = v;
        }
      }
    }
  }
}

// ---------------------------------------------------------------------------
// Block-cooperative causal flash attention.
// Block = 8 waves = one (b,h) and 8 consecutive 16-row q-tiles.
// K and V^T tiles (32 keys) staged to LDS with async copies, double-buffered;
// all fragment reads come from LDS (contiguous ds_load_b128 pairs).
// ---------------------------------------------------------------------------
#define NEG_BIG (-3.0e38f)

__global__ __launch_bounds__(256)
void flash_attn_wmma(const __bf16* __restrict__ Qh, const __bf16* __restrict__ Kh,
                     const __bf16* __restrict__ Vt, __bf16* __restrict__ Ob) {
  __shared__ __align__(32) __bf16 kt_lds[2][32 * 64];  // [key][d]
  __shared__ __align__(32) __bf16 vt_lds[2][64 * 32];  // [d][key]
  __shared__ __align__(32) __bf16 plds[8][16 * 32];    // per-wave P staging

  const int tid  = threadIdx.x;
  const int lane = tid & 31;
  const int l16  = lane & 15;
  const int half = lane >> 4;
  const int wib  = tid >> 5;

  const int bh = blockIdx.x >> 4;       // 64 (b,h) pairs
  const int qg = blockIdx.x & 15;       // q-tile group of 8
  const int qt = qg * 8 + wib;          // this wave's q-tile

  const __bf16* Qp  = Qh + (size_t)bh * 2048 * 64;
  const __bf16* Kp  = Kh + (size_t)bh * 2048 * 64;
  const __bf16* Vtp = Vt + (size_t)bh * 64 * 2048;

  // Q A-fragments (K-dim 0..31 / 32..63), direct from global
  v16bf qa0, qa1;
  {
    const __bf16* qrow = Qp + (size_t)(qt * 16 + l16) * 64;
    v8bf x0 = *(const v8bf*)(qrow + half * 16);
    v8bf x1 = *(const v8bf*)(qrow + half * 16 + 8);
    v8bf x2 = *(const v8bf*)(qrow + 32 + half * 16);
    v8bf x3 = *(const v8bf*)(qrow + 40 + half * 16);
#pragma unroll
    for (int i = 0; i < 8; ++i) {
      qa0[i] = x0[i]; qa0[8 + i] = x1[i];
      qa1[i] = x2[i]; qa1[8 + i] = x3[i];
    }
  }

  float mrow[8], lrow[8];
  v8f acc[4] = {v8f{}, v8f{}, v8f{}, v8f{}};
#pragma unroll
  for (int r = 0; r < 8; ++r) { mrow[r] = NEG_BIG; lrow[r] = 0.0f; }

  const int rowg_base = qt * 16 + half * 8;
  const int nkt_w = (qt * 16 + 15) / 32 + 1;             // this wave's causal bound
  const int nkt_b = ((qg * 8 + 7) * 16 + 15) / 32 + 1;   // block-uniform bound

  // Cooperative stagers: 256 threads x 16B = one 4KB tile per call.
  auto stageK = [&](int kt, int buf) {   // K tile rows are globally contiguous
    copy16B_g2l(Kp + (size_t)kt * 32 * 64 + tid * 8, &kt_lds[buf][tid * 8]);
  };
  auto stageV = [&](int kt, int buf) {   // V^T rows: stride 2048, 32 keys wide
    const int d = tid >> 2, q = tid & 3;
    copy16B_g2l(Vtp + (size_t)d * 2048 + kt * 32 + q * 8,
                &vt_lds[buf][d * 32 + q * 8]);
  };

  stageK(0, 0);
  stageV(0, 0);
  wait_stage_and_sync();

  for (int kt = 0; kt < nkt_b; ++kt) {
    const int cur = kt & 1;
    if (kt + 1 < nkt_b) {               // overlap next tile's copy with compute
      stageK(kt + 1, cur ^ 1);
      stageV(kt + 1, cur ^ 1);
    }

    if (kt < nkt_w) {                   // wave-uniform causal guard
      // ---- S = Q K^T (16x32 scores: 2 col-tiles x 2 k-steps) ----
      v8f s0 = v8f{}, s1 = v8f{};
#pragma unroll
      for (int cgrp = 0; cgrp < 2; ++cgrp) {
        const __bf16* krow = &kt_lds[cur][(cgrp * 16 + l16) * 64];
        v16bf b0 = *(const v16bf*)(krow + half * 16);
        v16bf b1 = *(const v16bf*)(krow + 32 + half * 16);
        if (cgrp == 0) { s0 = wmma_bf16(qa0, b0, s0); s0 = wmma_bf16(qa1, b1, s0); }
        else           { s1 = wmma_bf16(qa0, b0, s1); s1 = wmma_bf16(qa1, b1, s1); }
      }

      // ---- online softmax (rows live across 16 lanes of each half) ----
      float p0[8], p1[8];
      const int c0g = kt * 32 + l16;
      const int c1g = c0g + 16;
#pragma unroll
      for (int r = 0; r < 8; ++r) {
        const int rowg = rowg_base + r;
        float v0 = (c0g <= rowg) ? s0[r] * 0.125f : NEG_BIG;
        float v1 = (c1g <= rowg) ? s1[r] * 0.125f : NEG_BIG;
        float t = fmaxf(v0, v1);
        t = fmaxf(t, __shfl_xor(t, 1, 32));
        t = fmaxf(t, __shfl_xor(t, 2, 32));
        t = fmaxf(t, __shfl_xor(t, 4, 32));
        t = fmaxf(t, __shfl_xor(t, 8, 32));
        const float m2 = fmaxf(mrow[r], t);
        const float e0 = __expf(v0 - m2);
        const float e1 = __expf(v1 - m2);
        float srt = e0 + e1;
        srt += __shfl_xor(srt, 1, 32);
        srt += __shfl_xor(srt, 2, 32);
        srt += __shfl_xor(srt, 4, 32);
        srt += __shfl_xor(srt, 8, 32);
        const float f = __expf(mrow[r] - m2);
        lrow[r] = lrow[r] * f + srt;
        mrow[r] = m2;
        p0[r] = e0; p1[r] = e1;
        acc[0][r] *= f; acc[1][r] *= f; acc[2][r] *= f; acc[3][r] *= f;
      }

      // ---- P: C-layout -> A-layout via this wave's LDS tile ----
      __bf16* pl = plds[wib];
#pragma unroll
      for (int r = 0; r < 8; ++r) {
        const int prow = half * 8 + r;
        pl[prow * 32 + l16]      = (__bf16)p0[r];
        pl[prow * 32 + 16 + l16] = (__bf16)p1[r];
      }
      asm volatile("s_wait_dscnt 0" ::: "memory");
      v16bf pa;
      {
        v8bf lo = *(const v8bf*)(pl + l16 * 32 + half * 16);
        v8bf hi = *(const v8bf*)(pl + l16 * 32 + half * 16 + 8);
#pragma unroll
        for (int i = 0; i < 8; ++i) { pa[i] = lo[i]; pa[8 + i] = hi[i]; }
      }

      // ---- O += P V : V^T tile makes B-fragments contiguous in LDS ----
#pragma unroll
      for (int t = 0; t < 4; ++t) {
        v16bf bv = *(const v16bf*)(&vt_lds[cur][(t * 16 + l16) * 32 + half * 16]);
        acc[t] = wmma_bf16(pa, bv, acc[t]);
      }
    }

    wait_stage_and_sync();              // staging done + all reads of cur done
  }

  // normalize, store bf16 in [b, s, h*64+d] for the output projection
  const int b = bh >> 4, h = bh & 15;
#pragma unroll
  for (int r = 0; r < 8; ++r) {
    const float inv = 1.0f / lrow[r];
    const int s = qt * 16 + half * 8 + r;
    const size_t base = ((size_t)b * 2048 + s) * 1024 + h * 64;
#pragma unroll
    for (int t = 0; t < 4; ++t)
      Ob[base + t * 16 + l16] = (__bf16)(acc[t][r] * inv);
  }
}

// ---------------------------------------------------------------------------
// Host launcher
// ---------------------------------------------------------------------------
extern "C" void kernel_launch(void* const* d_in, const int* in_sizes, int n_in,
                              void* d_out, int out_size, void* d_ws, size_t ws_size,
                              hipStream_t stream) {
  const float* x  = (const float*)d_in[0];
  const float* Wq = (const float*)d_in[1];
  const float* Wk = (const float*)d_in[2];
  const float* Wv = (const float*)d_in[3];
  const float* Wo = (const float*)d_in[4];

  const int B = 4, S = 2048, D = 1024;
  const int M = B * S;                    // 8192
  const size_t nx = (size_t)M * D;        // 8,388,608
  const size_t nw = (size_t)D * D;        // 1,048,576

  char* p = (char*)d_ws;
  __bf16* xb  = (__bf16*)p; p += nx * 2;
  __bf16* wqb = (__bf16*)p; p += nw * 2;
  __bf16* wkb = (__bf16*)p; p += nw * 2;
  __bf16* wvb = (__bf16*)p; p += nw * 2;
  __bf16* wob = (__bf16*)p; p += nw * 2;
  __bf16* Qhh = (__bf16*)p; p += nx * 2;  // [B,H,S,Dk]
  __bf16* Khh = (__bf16*)p; p += nx * 2;  // [B,H,S,Dk]
  __bf16* Vtt = (__bf16*)p; p += nx * 2;  // [B,H,Dk,S]  (transposed)
  __bf16* Obf = (__bf16*)p; p += nx * 2;  // [B,S,D]

  cvt_f32_to_bf16<<<(int)((nx + 255) / 256), 256, 0, stream>>>(x,  xb,  (int)nx);
  cvt_f32_to_bf16<<<(int)((nw + 255) / 256), 256, 0, stream>>>(Wq, wqb, (int)nw);
  cvt_f32_to_bf16<<<(int)((nw + 255) / 256), 256, 0, stream>>>(Wk, wkb, (int)nw);
  cvt_f32_to_bf16<<<(int)((nw + 255) / 256), 256, 0, stream>>>(Wv, wvb, (int)nw);
  cvt_f32_to_bf16<<<(int)((nw + 255) / 256), 256, 0, stream>>>(Wo, wob, (int)nw);

  // 64x64 tiles: (M/64)*(D/64) = 2048 wave-jobs, 8 per block -> 256 blocks
  const int gemm_blocks = ((M / 64) * (D / 64) + 7) / 8;
  gemm_bf16_wmma<<<gemm_blocks, 256, 0, stream>>>(xb, wqb, Qhh, M, D, D, 0);
  gemm_bf16_wmma<<<gemm_blocks, 256, 0, stream>>>(xb, wkb, Khh, M, D, D, 0);
  gemm_bf16_wmma<<<gemm_blocks, 256, 0, stream>>>(xb, wvb, Vtt, M, D, D, 2);

  // 64 (b,h) x 16 q-groups = 1024 blocks
  flash_attn_wmma<<<1024, 256, 0, stream>>>(Qhh, Khh, Vtt, Obf);

  gemm_bf16_wmma<<<gemm_blocks, 256, 0, stream>>>(Obf, wob, d_out, M, D, D, 1);
}
